// LSTMClassifier_56126632624387
// MI455X (gfx1250) — compile-verified, matrix operand order checked by reading, not run on hardware
//
#include <hip/hip_runtime.h>

// ---------------------------------------------------------------------------
// LSTM classifier for MI455X (gfx1250, wave32, WMMA 16x16x32 f16 -> f32 acc)
//
// One wave (32 lanes) owns a 16-sample batch tile. The full hidden sequence
// [T=111][16][H=32] is kept in LDS as f16 and processed layer-by-layer in
// place. Each timestep/layer = 16x v_wmma_f32_16x16x32_f16 (8 gate N-tiles
// x {input, recurrent}). Attention + MLP head fused in the same kernel.
// All nonlinearities are branchless v_exp/v_rcp sequences.
// ---------------------------------------------------------------------------

#define T_SEQ  111
#define EMB_D  32
#define HID_D  32
#define NLAYER 4
#define BTILE  16

typedef __attribute__((ext_vector_type(16))) _Float16 v16h;
typedef __attribute__((ext_vector_type(8)))  _Float16 v8h;
typedef __attribute__((ext_vector_type(8)))  float    v8f;

// Branchless sigmoid: v_exp_f32 + v_rcp_f32, no libm, no divergence.
__device__ __forceinline__ float fast_sigmoid(float x) {
    return __builtin_amdgcn_rcpf(1.0f + __expf(-x));
}

// Branchless tanh: 1 - 2/(exp(2x)+1). Saturates to +/-1 at both infinities
// (rcp(inf) == 0), so no clamping or special-casing needed.
__device__ __forceinline__ float fast_tanh(float x) {
    return 1.0f - 2.0f * __builtin_amdgcn_rcpf(__expf(2.0f * x) + 1.0f);
}

__device__ __forceinline__ v8f splat8(float s) {
    v8f r;
#pragma unroll
    for (int i = 0; i < 8; ++i) r[i] = s;
    return r;
}

__device__ __forceinline__ v16h zero16h() {
    v16h r;
#pragma unroll
    for (int i = 0; i < 16; ++i) r[i] = (_Float16)0.0f;
    return r;
}

// Gather a 16x32 f16 A-operand tile from an LDS row (plain [m][k] layout).
// Lane L holds row m = L&15, K slices {kb..kb+7, kb+16..kb+23}, kb = 8*(L>>4).
__device__ __forceinline__ v16h pack_a_lds(const _Float16* row, int kbase) {
    v8h lo = *(const v8h*)(row + kbase);
    v8h hi = *(const v8h*)(row + kbase + 16);
    return __builtin_shufflevector(lo, hi, 0, 1, 2, 3, 4, 5, 6, 7,
                                           8, 9, 10, 11, 12, 13, 14, 15);
}

// Load a B-operand tile (gate rows of W, f32 in global) into f16 B layout.
// Lane L of tile nt holds gate row g = nt*16 + (L&15), same K slice pattern.
__device__ __forceinline__ v16h load_b_tile(const float* __restrict__ Wbase,
                                            int g, int kbase) {
    const float* p = Wbase + g * 32 + kbase;
    v16h r;
#pragma unroll
    for (int i = 0; i < 8; ++i) r[i] = (_Float16)p[i];
#pragma unroll
    for (int i = 0; i < 8; ++i) r[8 + i] = (_Float16)p[16 + i];
    return r;
}

__global__ __launch_bounds__(32) void lstm_cls_kernel(
    const int*   __restrict__ x,    // [B][T]
    const float* __restrict__ emb,  // [VOCAB][32]
    const float* __restrict__ Wih,  // [4][128][32]
    const float* __restrict__ Whh,  // [4][128][32]
    const float* __restrict__ bih,  // [4][128]
    const float* __restrict__ bhh,  // [4][128]
    const float* __restrict__ Wa,   // [32]
    const float* __restrict__ ba,   // [1]
    const float* __restrict__ W1,   // [32][32]
    const float* __restrict__ b1,   // [32]
    const float* __restrict__ W2,   // [32][3]
    const float* __restrict__ b2,   // [3]
    float*       __restrict__ out)  // [B][3]
{
    __shared__ __align__(16) _Float16 s_buf[T_SEQ][BTILE][HID_D]; // 113664 B
    __shared__ float s_score[T_SEQ][BTILE];                       //   7104 B
    __shared__ float s_ms[2][BTILE];                              //    128 B
    __shared__ __align__(16) float s_ctx[BTILE][HID_D];           //   2048 B
    __shared__ float s_hid[BTILE][HID_D];                         //   2048 B

    const int lane  = threadIdx.x;
    const int lrow  = lane & 15;
    const int khalf = lane >> 4;
    const int kbase = khalf * 8;
    const int b0    = blockIdx.x * BTILE;

    // ---------------- Phase 0: embedding gather -> LDS f16 ----------------
    for (int idx = lane; idx < T_SEQ * BTILE; idx += 32) {
        const int t = idx >> 4;
        const int m = idx & 15;
        const int tok = x[(b0 + m) * T_SEQ + t];
        const float4* er = (const float4*)(emb + (long)tok * EMB_D);
        _Float16* dst = &s_buf[t][m][0];
#pragma unroll
        for (int q = 0; q < 8; ++q) {
            float4 v = er[q];
            dst[q * 4 + 0] = (_Float16)v.x;
            dst[q * 4 + 1] = (_Float16)v.y;
            dst[q * 4 + 2] = (_Float16)v.z;
            dst[q * 4 + 3] = (_Float16)v.w;
        }
    }
    __syncthreads();

    // ---------------- Phase 1: 4 LSTM layers, in place in LDS --------------
    for (int l = 0; l < NLAYER; ++l) {
        const float* wih_l = Wih + l * 4 * HID_D * EMB_D;
        const float* whh_l = Whh + l * 4 * HID_D * HID_D;

        if (l + 1 < NLAYER && lane == 0) {
            __builtin_prefetch(Wih + (l + 1) * 4 * HID_D * EMB_D, 0, 1);
            __builtin_prefetch(Whh + (l + 1) * 4 * HID_D * HID_D, 0, 1);
        }

        // Stage this layer's weights in registers (B layout, f16).
        v16h Bih[8], Bhh[8];
        float bias[8];
#pragma unroll
        for (int nt = 0; nt < 8; ++nt) {
            const int g = nt * 16 + lrow;
            Bih[nt] = load_b_tile(wih_l, g, kbase);
            Bhh[nt] = load_b_tile(whh_l, g, kbase);
            bias[nt] = bih[l * 128 + g] + bhh[l * 128 + g];
        }

        v16h a_h = zero16h();
        v8f  cst[2];
        cst[0] = splat8(0.0f);
        cst[1] = splat8(0.0f);

        for (int t = 0; t < T_SEQ; ++t) {
            // Layer input at time t (A layout from LDS, two ds_load_b128).
            const v16h a_x = pack_a_lds(&s_buf[t][lrow][0], kbase);

            // Gate tiles: i -> {0,1}, f -> {2,3}, g -> {4,5}, o -> {6,7};
            // half hf covers hidden columns [hf*16, hf*16+16).
#pragma unroll
            for (int hf = 0; hf < 2; ++hf) {
                v8f zi = splat8(bias[0 + hf]);
                zi = __builtin_amdgcn_wmma_f32_16x16x32_f16(false, a_x, false, Bih[0 + hf], (short)0, zi, false, false);
                zi = __builtin_amdgcn_wmma_f32_16x16x32_f16(false, a_h, false, Bhh[0 + hf], (short)0, zi, false, false);
                v8f zf = splat8(bias[2 + hf]);
                zf = __builtin_amdgcn_wmma_f32_16x16x32_f16(false, a_x, false, Bih[2 + hf], (short)0, zf, false, false);
                zf = __builtin_amdgcn_wmma_f32_16x16x32_f16(false, a_h, false, Bhh[2 + hf], (short)0, zf, false, false);
                v8f zg = splat8(bias[4 + hf]);
                zg = __builtin_amdgcn_wmma_f32_16x16x32_f16(false, a_x, false, Bih[4 + hf], (short)0, zg, false, false);
                zg = __builtin_amdgcn_wmma_f32_16x16x32_f16(false, a_h, false, Bhh[4 + hf], (short)0, zg, false, false);
                v8f zo = splat8(bias[6 + hf]);
                zo = __builtin_amdgcn_wmma_f32_16x16x32_f16(false, a_x, false, Bih[6 + hf], (short)0, zo, false, false);
                zo = __builtin_amdgcn_wmma_f32_16x16x32_f16(false, a_h, false, Bhh[6 + hf], (short)0, zo, false, false);

                // Elementwise cell update in C/D layout:
                // VGPR v, lane L -> (m = v + 8*(L>>4), n = hf*16 + (L&15)).
#pragma unroll
                for (int v = 0; v < 8; ++v) {
                    const float iv = fast_sigmoid(zi[v]);
                    const float fv = fast_sigmoid(zf[v]);
                    const float gv = fast_tanh(zg[v]);
                    const float ov = fast_sigmoid(zo[v]);
                    const float cv = fv * cst[hf][v] + iv * gv;
                    cst[hf][v] = cv;
                    const float hv = ov * fast_tanh(cv);
                    s_buf[t][v + 8 * khalf][hf * 16 + lrow] = (_Float16)hv;
                }
            }

            // Re-gather the freshly written h_t as next step's A operand.
            a_h = pack_a_lds(&s_buf[t][lrow][0], kbase);
        }
        __syncthreads();
    }

    // ---------------- Phase 2: attention over time ----------------
    float wa[32];
#pragma unroll
    for (int k = 0; k < 32; ++k) wa[k] = Wa[k];
    const float bav = ba[0];

    for (int idx = lane; idx < T_SEQ * BTILE; idx += 32) {
        const int t = idx >> 4;
        const int m = idx & 15;
        float s = bav;
#pragma unroll
        for (int k = 0; k < 32; ++k) s += (float)s_buf[t][m][k] * wa[k];
        s_score[t][m] = s;
    }
    __syncthreads();

    if (lane < 16) {
        float mx = -1e30f;
        for (int t = 0; t < T_SEQ; ++t) mx = fmaxf(mx, s_score[t][lane]);
        float se = 0.0f;
        for (int t = 0; t < T_SEQ; ++t) se += __expf(s_score[t][lane] - mx);
        s_ms[0][lane] = mx;
        s_ms[1][lane] = se;
    }
    __syncthreads();

    {
        const int m  = lrow;
        const int k0 = khalf * 16;
        const float mx  = s_ms[0][m];
        const float inv = __builtin_amdgcn_rcpf(s_ms[1][m]);
        float ctx[16];
#pragma unroll
        for (int j = 0; j < 16; ++j) ctx[j] = 0.0f;
        for (int t = 0; t < T_SEQ; ++t) {
            const float w = __expf(s_score[t][m] - mx) * inv;
#pragma unroll
            for (int j = 0; j < 16; ++j)
                ctx[j] += w * (float)s_buf[t][m][k0 + j];
        }
#pragma unroll
        for (int j = 0; j < 16; ++j) s_ctx[m][k0 + j] = ctx[j];
    }
    __syncthreads();

    // ---------------- Phase 3: MLP head (ctx @ W1 via WMMA) ----------------
    v16h a_c;
#pragma unroll
    for (int i = 0; i < 8; ++i) {
        a_c[i]     = (_Float16)s_ctx[lrow][kbase + i];
        a_c[8 + i] = (_Float16)s_ctx[lrow][kbase + 16 + i];
    }
#pragma unroll
    for (int nt = 0; nt < 2; ++nt) {
        const int n = nt * 16 + lrow;
        v16h bw;
#pragma unroll
        for (int i = 0; i < 8; ++i) {
            bw[i]     = (_Float16)W1[(kbase + i) * 32 + n];
            bw[8 + i] = (_Float16)W1[(kbase + 16 + i) * 32 + n];
        }
        v8f acc = splat8(b1[n]);
        acc = __builtin_amdgcn_wmma_f32_16x16x32_f16(false, a_c, false, bw, (short)0, acc, false, false);
#pragma unroll
        for (int v = 0; v < 8; ++v)
            s_hid[v + 8 * khalf][n] = fast_tanh(acc[v]);
    }
    __syncthreads();

    if (lane < 16) {
        const int m = lane;
#pragma unroll
        for (int j = 0; j < 3; ++j) {
            float o = b2[j];
#pragma unroll
            for (int k = 0; k < 32; ++k) o += s_hid[m][k] * W2[k * 3 + j];
            out[(b0 + m) * 3 + j] = o;
        }
    }
}

extern "C" void kernel_launch(void* const* d_in, const int* in_sizes, int n_in,
                              void* d_out, int out_size, void* d_ws, size_t ws_size,
                              hipStream_t stream) {
    const int*   x   = (const int*)  d_in[0];
    const float* emb = (const float*)d_in[1];
    const float* Wih = (const float*)d_in[2];
    const float* Whh = (const float*)d_in[3];
    const float* bih = (const float*)d_in[4];
    const float* bhh = (const float*)d_in[5];
    const float* Wa  = (const float*)d_in[6];
    const float* ba  = (const float*)d_in[7];
    const float* W1  = (const float*)d_in[8];
    const float* b1  = (const float*)d_in[9];
    const float* W2  = (const float*)d_in[10];
    const float* b2  = (const float*)d_in[11];

    const int batch = in_sizes[0] / T_SEQ;   // 8192
    const int nblk  = batch / BTILE;         // 512

    lstm_cls_kernel<<<dim3(nblk), dim3(32), 0, stream>>>(
        x, emb, Wih, Whh, bih, bhh, Wa, ba, W1, b1, W2, b2, (float*)d_out);
}
